// SparseEncoderIN_61177514165064
// MI455X (gfx1250) — matrix operand order repaired and checked by reading, not compiled, not run on hardware
//
#include <hip/hip_runtime.h>
#include <hip/hip_bf16.h>
#include <stdint.h>
#include <stddef.h>

// ---------------------------------------------------------------------------
// MI455X (gfx1250) sparse-3D-CNN encoder.
//   conv3x3x3 (implicit GEMM, bf16 WMMA, f32 acc) -> LayerNorm(C) -> LeakyReLU
//   -> mask, 4 stages, then spatial max.
// Precision: bf16 activations/weights (halves HBM traffic vs fp32; WMMA-native),
// fp32 accumulate + fp32 LayerNorm statistics.
// Taps fully unrolled: kd/kh/kw are compile-time, so per-tap addresses fold
// into load immediate offsets and validity tests are 3+3+3 precomputed bools.
// ---------------------------------------------------------------------------

typedef __attribute__((ext_vector_type(16))) __bf16 v16bf;
typedef __attribute__((ext_vector_type(8)))  float  v8f;

union ABFrag { v16bf v; uint4 q[2]; };

__device__ __forceinline__ unsigned short f2bf(float f) {
  unsigned u = __float_as_uint(f);
  u += 0x7fffu + ((u >> 16) & 1u);           // round-to-nearest-even
  return (unsigned short)(u >> 16);
}

// ---- Stage 0: NCDHW fp32 * mask -> NDHWC bf16, Cin 3 padded to 32 ----------
__global__ void prep_kernel(const float* __restrict__ x, const int* __restrict__ mask,
                            unsigned short* __restrict__ act0, float* __restrict__ m0,
                            int total) {
  int vi = blockIdx.x * blockDim.x + threadIdx.x;
  if (vi >= total) return;
  int b  = vi >> 18;               // 64^3 = 262144 voxels / image
  int sp = vi & 262143;
  float m = (mask[vi] != 0) ? 1.0f : 0.0f;
  m0[vi] = m;
  unsigned short row[32];
#pragma unroll
  for (int c = 0; c < 32; ++c) row[c] = 0;
#pragma unroll
  for (int c = 0; c < 3; ++c) row[c] = f2bf(x[(size_t)(b * 3 + c) * 262144 + sp] * m);
  uint4* dst = (uint4*)(act0 + (size_t)vi * 32);
  const uint4* src = (const uint4*)row;
#pragma unroll
  for (int q = 0; q < 4; ++q) dst[q] = src[q];
}

// ---- Weight pack: [Cout][Cin][27] fp32 -> per-lane WMMA B fragments (bf16) --
// Linear layout: ((((tap*KC + kc)*CT + ct)*32 + lane)*16 + j)
// lane = khalf*16 + col; element j <-> K = (j>>3)*16 + khalf*8 + (j&7) (ISA 16-bit B layout).
__global__ void pack_w_kernel(const float* __restrict__ w, unsigned short* __restrict__ wp,
                              int Cin, int CPI, int COUT, int total) {
  int t = blockIdx.x * blockDim.x + threadIdx.x;
  if (t >= total) return;
  int idx  = t;
  int j    = idx & 15;  idx >>= 4;
  int lane = idx & 31;  idx >>= 5;
  int CT   = COUT >> 4;
  int ct   = idx % CT;  idx /= CT;
  int KC   = CPI >> 5;
  int kc   = idx % KC;  idx /= KC;
  int tap  = idx;                              // kd*9 + kh*3 + kw
  int cin  = kc * 32 + ((j >> 3) << 4) + ((lane >> 4) << 3) + (j & 7);
  int cout = ct * 16 + (lane & 15);
  float val = (cin < Cin) ? w[((size_t)cout * Cin + cin) * 27 + tap] : 0.0f;
  wp[t] = f2bf(val);
}

// ---- Fused conv (implicit GEMM, WMMA) + LayerNorm(C) + LeakyReLU + mask -----
template <int CPI, int COUT, int S, int DI, int DO, bool OUTF32>
__global__ __launch_bounds__(128) void conv_block(
    const unsigned short* __restrict__ act_in,   // [B][DI][DI][DI][CPI] bf16
    const unsigned short* __restrict__ wp,       // packed weights bf16
    const float* __restrict__ gamma, const float* __restrict__ beta,
    const float* __restrict__ mask_in,           // [B][DI][DI][DI]
    float* __restrict__ mask_out,                // [B][DO][DO][DO]
    void* __restrict__ outp)                     // [B][DO][DO][DO][COUT]
{
  constexpr int KC = CPI / 32;     // k-steps per tap
  constexpr int CT = COUT / 16;    // Cout tiles
  constexpr int NW = 4;            // waves / workgroup (128 thr = 4 x wave32)
  constexpr int NT = CT / NW;      // Cout tiles per wave
  __shared__ float smem[16 * COUT];

  const int lane = threadIdx.x & 31;
  const int wave = threadIdx.x >> 5;
  const int tileBase = blockIdx.x * 16;

  // This lane's A-matrix row (voxel) + K-half, per ISA 16-bit A layout.
  const int m     = lane & 15;
  const int khalf = lane >> 4;
  int vi = tileBase + m;
  int wo = vi % DO; int tt = vi / DO;
  int ho = tt % DO; tt /= DO;
  int dd = tt % DO;
  int bb = tt / DO;

  // Per-dimension validity, computed once (tap loop indexes with constants).
  const int id0 = dd * S - 1, ih0 = ho * S - 1, iw0 = wo * S - 1;
  bool dok[3], hok[3], wok[3];
#pragma unroll
  for (int k = 0; k < 3; ++k) {
    dok[k] = (unsigned)(id0 + k) < (unsigned)DI;
    hok[k] = (unsigned)(ih0 + k) < (unsigned)DI;
    wok[k] = (unsigned)(iw0 + k) < (unsigned)DI;
  }
  // Lane base pointer at tap (0,0,0), channel-half folded in.
  const unsigned short* pBase =
      act_in + ((((ptrdiff_t)bb * DI + id0) * DI + ih0) * DI + iw0) * CPI + khalf * 8;

  v8f acc[NT];
#pragma unroll
  for (int t = 0; t < NT; ++t)
#pragma unroll
    for (int r = 0; r < 8; ++r) acc[t][r] = 0.0f;

#pragma unroll
  for (int kd = 0; kd < 3; ++kd) {
#pragma unroll
    for (int kh = 0; kh < 3; ++kh) {
#pragma unroll
      for (int kw = 0; kw < 3; ++kw) {
        constexpr int DI2 = DI * DI;   // silence none; just clarity
        const int tap = (kd * 3 + kh) * 3 + kw;
        const bool ok = dok[kd] && hok[kh] && wok[kw];
        const unsigned short* pA =
            pBase + ((ptrdiff_t)kd * DI2 + kh * DI + kw) * CPI;
        if (tap + 1 < 27)   // warm L2/WGP$ for the next tap's weight fragments
          __builtin_prefetch(wp + ((size_t)(tap + 1) * KC * CT) * 512 + (size_t)lane * 16,
                             0, 1);
#pragma unroll 1
        for (int kc = 0; kc < KC; ++kc) {
          ABFrag a;
          if (ok) {
            const uint4* p = (const uint4*)(pA + kc * 32);
            a.q[0] = p[0];        // K = khalf*8 + 0..7
            a.q[1] = p[2];        // K = 16 + khalf*8 + 0..7
          } else {
            a.q[0] = make_uint4(0, 0, 0, 0);
            a.q[1] = make_uint4(0, 0, 0, 0);
          }
          const unsigned short* pBb =
              wp + (((size_t)tap * KC + kc) * CT) * 512 + (size_t)lane * 16;
#pragma unroll
          for (int t = 0; t < NT; ++t) {
            const int ct = wave + t * NW;
            ABFrag bf;
            const uint4* pb = (const uint4*)(pBb + (size_t)ct * 512);
            bf.q[0] = pb[0];
            bf.q[1] = pb[1];
            acc[t] = __builtin_amdgcn_wmma_f32_16x16x32_bf16(
                false, a.v, false, bf.v, (short)0, acc[t], false, false);
          }
        }
      }
    }
  }

  // Spill accumulators to LDS: C/D layout -> (voxel = r + 8*khalf, cout).
#pragma unroll
  for (int t = 0; t < NT; ++t) {
    const int ct = wave + t * NW;
#pragma unroll
    for (int r = 0; r < 8; ++r)
      smem[(r + (khalf << 3)) * COUT + ct * 16 + m] = acc[t][r];
  }
  __syncthreads();

  // LayerNorm over channels + LeakyReLU + mask; each wave owns 4 voxels.
  constexpr int CPL = COUT / 32;
#pragma unroll 1
  for (int vloc = 0; vloc < 4; ++vloc) {
    const int v   = wave * 4 + vloc;
    const int vig = tileBase + v;
    int wo2 = vig % DO; int u = vig / DO;
    int ho2 = u % DO;  u /= DO;
    int dd2 = u % DO;
    int b2  = u / DO;

    float mout;
    if (S == 1) {
      mout = mask_in[vig];
    } else {  // strided SparseConv3d: active iff any input in 3x3x3 window active
      const int jd0 = dd2 * S - 1, jh0 = ho2 * S - 1, jw0 = wo2 * S - 1;
      const float* mBase =
          mask_in + (((ptrdiff_t)b2 * DI + jd0) * DI + jh0) * DI + jw0;
      float mm = 0.f;
#pragma unroll
      for (int kd = 0; kd < 3; ++kd)
#pragma unroll
        for (int kh = 0; kh < 3; ++kh)
#pragma unroll
          for (int kw = 0; kw < 3; ++kw) {
            const bool okm = (unsigned)(jd0 + kd) < (unsigned)DI &&
                             (unsigned)(jh0 + kh) < (unsigned)DI &&
                             (unsigned)(jw0 + kw) < (unsigned)DI;
            if (okm)
              mm = fmaxf(mm, mBase[((ptrdiff_t)kd * DI + kh) * DI + kw]);
          }
      mout = (mm > 0.f) ? 1.f : 0.f;
    }

    float s = 0.f, s2 = 0.f;
#pragma unroll
    for (int k = 0; k < CPL; ++k) {
      float y = smem[v * COUT + lane + 32 * k];
      s += y; s2 += y * y;
    }
#pragma unroll
    for (int off = 16; off >= 1; off >>= 1) {   // wave32 butterfly
      s  += __shfl_xor(s,  off, 32);
      s2 += __shfl_xor(s2, off, 32);
    }
    const float mu  = s / COUT;
    const float var = s2 / COUT - mu * mu;
    const float rs  = rsqrtf(var + 1e-5f);

#pragma unroll
    for (int k = 0; k < CPL; ++k) {
      const int c = lane + 32 * k;
      float y = smem[v * COUT + c];
      float o = (y - mu) * rs * gamma[c] + beta[c];
      o = (o > 0.f) ? o : 0.2f * o;
      o *= mout;
      if (OUTF32) ((float*)outp)[(size_t)vig * COUT + c] = o;
      else        ((unsigned short*)outp)[(size_t)vig * COUT + c] = f2bf(o);
    }
    if (lane == 0) mask_out[vig] = mout;
  }
}

// ---- Final spatial max: [2][512 vox][512 ch] -> [2][512] --------------------
__global__ void maxred_kernel(const float* __restrict__ y, float* __restrict__ out) {
  int t = blockIdx.x * blockDim.x + threadIdx.x;
  if (t >= 1024) return;
  int b = t >> 9, c = t & 511;
  const float* p = y + (size_t)b * 512 * 512 + c;
  float m = -3.4e38f;
  for (int v = 0; v < 512; ++v) m = fmaxf(m, p[(size_t)v * 512]);
  out[t] = m;
}

extern "C" void kernel_launch(void* const* d_in, const int* in_sizes, int n_in,
                              void* d_out, int out_size, void* d_ws, size_t ws_size,
                              hipStream_t stream) {
  (void)in_sizes; (void)n_in; (void)out_size; (void)ws_size;
  const float* x    = (const float*)d_in[0];
  const int*   mask = (const int*)d_in[1];
  const float* w1 = (const float*)d_in[2];
  const float* g1 = (const float*)d_in[3];
  const float* b1 = (const float*)d_in[4];
  const float* w2 = (const float*)d_in[5];
  const float* g2 = (const float*)d_in[6];
  const float* b2 = (const float*)d_in[7];
  const float* w3 = (const float*)d_in[8];
  const float* g3 = (const float*)d_in[9];
  const float* b3 = (const float*)d_in[10];
  const float* w4 = (const float*)d_in[11];
  const float* g4 = (const float*)d_in[12];
  const float* b4 = (const float*)d_in[13];

  char* ws = (char*)d_ws;
  size_t off = 0;
  auto alloc = [&](size_t bytes) {
    size_t o = off; off += (bytes + 255) & ~(size_t)255; return o;
  };

  // ~137 MB of workspace (bf16 activations keep this well below fp32 cost).
  unsigned short* act0 = (unsigned short*)(ws + alloc((size_t)2 * 262144 * 32 * 2));
  float*          m0   = (float*)(ws + alloc((size_t)2 * 262144 * 4));
  unsigned short* act1 = (unsigned short*)(ws + alloc((size_t)2 * 262144 * 64 * 2));
  float*          m1   = (float*)(ws + alloc((size_t)2 * 262144 * 4));
  unsigned short* act2 = (unsigned short*)(ws + alloc((size_t)2 * 32768 * 128 * 2));
  float*          m2   = (float*)(ws + alloc((size_t)2 * 32768 * 4));
  unsigned short* act3 = (unsigned short*)(ws + alloc((size_t)2 * 4096 * 256 * 2));
  float*          m3   = (float*)(ws + alloc((size_t)2 * 4096 * 4));
  float*          y4   = (float*)(ws + alloc((size_t)2 * 512 * 512 * 4));
  float*          m4   = (float*)(ws + alloc((size_t)2 * 512 * 4));
  unsigned short* wp1  = (unsigned short*)(ws + alloc((size_t)27 * 32 * 64 * 2));
  unsigned short* wp2  = (unsigned short*)(ws + alloc((size_t)27 * 64 * 128 * 2));
  unsigned short* wp3  = (unsigned short*)(ws + alloc((size_t)27 * 128 * 256 * 2));
  unsigned short* wp4  = (unsigned short*)(ws + alloc((size_t)27 * 256 * 512 * 2));

  // Stage 0: mask + pack input.
  prep_kernel<<<(524288 + 255) / 256, 256, 0, stream>>>(x, mask, act0, m0, 524288);

  // Pack weights into WMMA fragment order.
  {
    int t1 = 27 * 32 * 64;
    pack_w_kernel<<<(t1 + 255) / 256, 256, 0, stream>>>(w1, wp1, 3, 32, 64, t1);
    int t2 = 27 * 64 * 128;
    pack_w_kernel<<<(t2 + 255) / 256, 256, 0, stream>>>(w2, wp2, 64, 64, 128, t2);
    int t3 = 27 * 128 * 256;
    pack_w_kernel<<<(t3 + 255) / 256, 256, 0, stream>>>(w3, wp3, 128, 128, 256, t3);
    int t4 = 27 * 256 * 512;
    pack_w_kernel<<<(t4 + 255) / 256, 256, 0, stream>>>(w4, wp4, 256, 256, 512, t4);
  }

  // Four fused conv+LN+LeakyReLU+mask stages.
  conv_block< 32,  64, 1, 64, 64, false><<<524288 / 16, 128, 0, stream>>>(
      act0, wp1, g1, b1, m0, m1, act1);
  conv_block< 64, 128, 2, 64, 32, false><<< 65536 / 16, 128, 0, stream>>>(
      act1, wp2, g2, b2, m1, m2, act2);
  conv_block<128, 256, 2, 32, 16, false><<<  8192 / 16, 128, 0, stream>>>(
      act2, wp3, g3, b3, m2, m3, act3);
  conv_block<256, 512, 2, 16,  8, true><<<   1024 / 16, 128, 0, stream>>>(
      act3, wp4, g4, b4, m3, m4, y4);

  // Spatial max -> [2, 512].
  maxred_kernel<<<4, 256, 0, stream>>>(y4, (float*)d_out);
}
